// FeatureAdaption_26259430048538
// MI455X (gfx1250) — compile-verified, matrix operand order checked by reading, not run on hardware
//
#include <hip/hip_runtime.h>

// ---------------- problem constants (from reference) ----------------
#define BATCH 4
#define CIN   128
#define COFF  64
#define COUT  128
#define HH    128
#define WW    128
#define PP    (HH * WW)          // 16384 pixels per batch image
#define KK    9                  // 3x3 taps
#define KDIM  (KK * CIN)         // 1152  (WMMA reduction length)
#define KBS   (KDIM / 32)        // 36 k-steps of 32 (f16 WMMA K)
#define NT    (COUT / 16)        // 8 n-tiles of 16 -> one per wave
#define ROWP  (KDIM + 16)        // padded LDS row stride (halves) to break bank conflicts

typedef _Float16 h16 __attribute__((ext_vector_type(16)));
typedef _Float16 h8  __attribute__((ext_vector_type(8)));
typedef _Float16 h4  __attribute__((ext_vector_type(4)));
typedef float    f8  __attribute__((ext_vector_type(8)));

static __device__ __forceinline__ int iclamp(int v, int lo, int hi) {
  return v < lo ? lo : (v > hi ? hi : v);
}

// ---------------- kernel 1: x NCHW -> NHWC (LDS-tiled transpose) ----------------
__global__ __launch_bounds__(1024)
void k_transpose(const float* __restrict__ x, float* __restrict__ xT) {
  __shared__ float t[32][33];
  const int b  = blockIdx.z;
  const int p0 = blockIdx.y * 32;
  const int c0 = blockIdx.x * 32;
  // coalesced read over p
  const int pr = p0 + threadIdx.x;
  const int cr = c0 + threadIdx.y;
  t[threadIdx.y][threadIdx.x] = x[((size_t)(b * CIN + cr)) * PP + pr];
  __syncthreads();
  // coalesced write over c
  const int pw = p0 + threadIdx.y;
  const int cw = c0 + threadIdx.x;
  xT[((size_t)(b * PP + pw)) * CIN + cw] = t[threadIdx.x][threadIdx.y];
}

// ---------------- kernel 2: 1x1 offset conv -> offT[pix][18] ----------------
__global__ __launch_bounds__(256)
void k_offsets(const float* __restrict__ off, const float* __restrict__ w_off,
               const float* __restrict__ b_off, float* __restrict__ offT) {
  const int idx = blockIdx.x * 256 + threadIdx.x;   // global pixel id, [0, B*PP)
  const int b = idx >> 14;
  const int p = idx & (PP - 1);
  const float* ob = off + (size_t)b * COFF * PP + p;
  float acc[18];
#pragma unroll
  for (int j = 0; j < 18; ++j) acc[j] = b_off[j];
  for (int c = 0; c < COFF; ++c) {
    const float v = ob[(size_t)c * PP];             // coalesced across threads
#pragma unroll
    for (int j = 0; j < 18; ++j) acc[j] += w_off[j * COFF + c] * v;
  }
  float* dst = offT + (size_t)idx * 18;
#pragma unroll
  for (int j = 0; j < 18; ++j) dst[j] = acc[j];
}

// ---------------- kernel 3: weight repack into WMMA B-fragment order ----------------
// layout: wpack[nt][kb][lane][slot] f16, slot = 16 halves per lane (32x16 f16 B frag)
// lane<16 : N=lane, slots 0..7 -> K 0..7,  slots 8..15 -> K 16..23
// lane>=16: N=lane-16, slots 0..7 -> K 8..15, slots 8..15 -> K 24..31
__global__ __launch_bounds__(256)
void k_wpack(const float* __restrict__ w_def, _Float16* __restrict__ wpack) {
  const int idx  = blockIdx.x * 256 + threadIdx.x;  // [0, NT*KBS*32*16)
  const int slot = idx & 15;
  const int lane = (idx >> 4) & 31;
  const int kb   = (idx >> 9) % KBS;
  const int nt   = idx / (512 * KBS);
  const int n    = lane & 15;
  const int hi   = lane >> 4;
  const int kloc = ((slot < 8) ? slot : (slot + 8)) + hi * 8;
  const int kdim = kb * 32 + kloc;                  // kdim = tap*128 + c
  const int o    = nt * 16 + n;
  const int tap  = kdim >> 7;
  const int c    = kdim & 127;
  wpack[idx] = (_Float16)w_def[((size_t)(o * CIN + c)) * KK + tap];
}

// ---------------- kernel 4: gather -> LDS f16 tile -> WMMA GEMM -> ReLU ----------------
__global__ __launch_bounds__(256)
void k_main(const float* __restrict__ xT, const float* __restrict__ offT,
            const _Float16* __restrict__ wpack, float* __restrict__ out) {
  __shared__ _Float16 vs[16 * ROWP];                // 16 pixels x 1152 (padded) f16 tile
  const int tid = threadIdx.x;
  const int tileBase = blockIdx.x * 16;             // global pixel of row 0 of tile
  const int b = tileBase >> 14;
  const float* xTb = xT + (size_t)b * PP * CIN;

  // ---- phase 1: deformable bilinear gather into LDS (f16) ----
  // unit = (pixel-in-tile m, tap, channel-quarter q): 16*9*4 = 576 units
  for (int u = tid; u < 16 * KK * 4; u += 256) {
    const int q   = u & 3;
    const int t9  = u >> 2;
    const int tap = t9 % KK;
    const int m   = t9 / KK;
    const int pix = tileBase + m;
    const int p   = pix & (PP - 1);
    const int h   = p >> 7;
    const int w   = p & (WW - 1);
    const float* ofp = offT + (size_t)pix * 18;
    const float dy = ofp[tap * 2 + 0];
    const float dx = ofp[tap * 2 + 1];
    const float ys = (float)(h - 1 + tap / 3) + dy;
    const float xs = (float)(w - 1 + tap % 3) + dx;
    const float y0f = floorf(ys), x0f = floorf(xs);
    const float ly = ys - y0f, lx = xs - x0f;
    const int iy0 = (int)y0f, ix0 = (int)x0f;
    const int iy1 = iy0 + 1,  ix1 = ix0 + 1;
    const float vy0 = (iy0 >= 0 && iy0 < HH) ? 1.f : 0.f;
    const float vy1 = (iy1 >= 0 && iy1 < HH) ? 1.f : 0.f;
    const float vx0 = (ix0 >= 0 && ix0 < WW) ? 1.f : 0.f;
    const float vx1 = (ix1 >= 0 && ix1 < WW) ? 1.f : 0.f;
    const float w00 = (1.f - ly) * (1.f - lx) * vy0 * vx0;
    const float w01 = (1.f - ly) * lx         * vy0 * vx1;
    const float w10 = ly         * (1.f - lx) * vy1 * vx0;
    const float w11 = ly         * lx         * vy1 * vx1;
    const int cy0 = iclamp(iy0, 0, HH - 1), cy1 = iclamp(iy1, 0, HH - 1);
    const int cx0 = iclamp(ix0, 0, WW - 1), cx1 = iclamp(ix1, 0, WW - 1);
    const float* r00 = xTb + ((size_t)(cy0 * WW + cx0)) * CIN;
    const float* r01 = xTb + ((size_t)(cy0 * WW + cx1)) * CIN;
    const float* r10 = xTb + ((size_t)(cy1 * WW + cx0)) * CIN;
    const float* r11 = xTb + ((size_t)(cy1 * WW + cx1)) * CIN;
    const int cBase = q * 32;
    _Float16* dst = vs + m * ROWP + tap * CIN + cBase;
#pragma unroll
    for (int cc = 0; cc < 8; ++cc) {
      const int c4 = cBase + cc * 4;
      const float4 a00 = *(const float4*)(r00 + c4);
      const float4 a01 = *(const float4*)(r01 + c4);
      const float4 a10 = *(const float4*)(r10 + c4);
      const float4 a11 = *(const float4*)(r11 + c4);
      h4 hv;
      hv.x = (_Float16)(w00 * a00.x + w01 * a01.x + w10 * a10.x + w11 * a11.x);
      hv.y = (_Float16)(w00 * a00.y + w01 * a01.y + w10 * a10.y + w11 * a11.y);
      hv.z = (_Float16)(w00 * a00.z + w01 * a01.z + w10 * a10.z + w11 * a11.z);
      hv.w = (_Float16)(w00 * a00.w + w01 * a01.w + w10 * a10.w + w11 * a11.w);
      *(h4*)(dst + cc * 4) = hv;                    // 8-byte LDS store
    }
  }
  __syncthreads();

  // ---- phase 2: 16x128 = A(16x1152,f16) x B(1152x128,f16), one 16-wide n-tile per wave ----
  const int lane = tid & 31;
  const int wv   = tid >> 5;                        // 0..7
  const int row  = lane & 15;                       // A: M = lane%16
  const int hi   = lane >> 4;                       // A: lanes 16-31 take K+8 octet
  const _Float16* aRow  = vs + row * ROWP + hi * 8;
  const _Float16* bBase = wpack + ((size_t)(wv * KBS) * 32 + lane) * 16;

  f8 acc = {0.f, 0.f, 0.f, 0.f, 0.f, 0.f, 0.f, 0.f};
#pragma unroll 4
  for (int kb = 0; kb < KBS; ++kb) {
    const _Float16* ap = aRow + kb * 32;
    const h8 alo = *(const h8*)ap;                  // K 0..7   (or 8..15 on hi lanes)
    const h8 ahi = *(const h8*)(ap + 16);           // K 16..23 (or 24..31 on hi lanes)
    const h16 a = __builtin_shufflevector(alo, ahi, 0,1,2,3,4,5,6,7,8,9,10,11,12,13,14,15);
    const _Float16* bp = bBase + (size_t)kb * (32 * 16);
    const h8 blo = *(const h8*)bp;
    const h8 bhi = *(const h8*)(bp + 8);
    const h16 bm = __builtin_shufflevector(blo, bhi, 0,1,2,3,4,5,6,7,8,9,10,11,12,13,14,15);
    acc = __builtin_amdgcn_wmma_f32_16x16x32_f16(
        /*neg_a=*/false, a, /*neg_b=*/false, bm,
        /*c_mod=*/(short)0, acc, /*reuse_a=*/false, /*reuse_b=*/false);
  }

  // ---- epilogue: ReLU + scatter to NCHW ----
  const int p0 = tileBase & (PP - 1);
  const int o  = wv * 16 + (lane & 15);
#pragma unroll
  for (int r = 0; r < 8; ++r) {
    const int M = r + hi * 8;                       // D layout: vgpr r, hi-lanes = M+8
    const float v = acc[r];
    out[((size_t)(b * COUT + o)) * PP + p0 + M] = v > 0.f ? v : 0.f;
  }
}

// ---------------- launcher ----------------
extern "C" void kernel_launch(void* const* d_in, const int* in_sizes, int n_in,
                              void* d_out, int out_size, void* d_ws, size_t ws_size,
                              hipStream_t stream) {
  (void)in_sizes; (void)n_in; (void)out_size; (void)ws_size;
  const float* x     = (const float*)d_in[0];   // [B,128,128,128]
  const float* off   = (const float*)d_in[1];   // [B,64,128,128]
  const float* w_off = (const float*)d_in[2];   // [18,64]
  const float* b_off = (const float*)d_in[3];   // [18]
  const float* w_def = (const float*)d_in[4];   // [128,128,3,3]
  float* out = (float*)d_out;                   // [B,128,128,128]

  char* ws = (char*)d_ws;
  float*    xT    = (float*)ws;                                   // 32 MB  NHWC x
  float*    offT  = (float*)(ws + (size_t)33554432);              // 4.5 MB [pix][18]
  _Float16* wpack = (_Float16*)(ws + (size_t)33554432 + 4718592); // 288 KB B frags

  k_transpose<<<dim3(CIN / 32, PP / 32, BATCH), dim3(32, 32, 1), 0, stream>>>(x, xT);
  k_offsets<<<(BATCH * PP) / 256, 256, 0, stream>>>(off, w_off, b_off, offT);
  k_wpack<<<(NT * KBS * 32 * 16) / 256, 256, 0, stream>>>(w_def, wpack);
  k_main<<<(BATCH * PP) / 16, 256, 0, stream>>>(xT, offT, wpack, out);
}
